// ContextDecoder_17162689315185
// MI455X (gfx1250) — matrix operand order repaired and checked by reading, not compiled, not run on hardware
//
#include <hip/hip_runtime.h>

// ---------------------------------------------------------------------------
// ContextDecoder for MI455X (gfx1250): WMMA f16->f32 GEMMs + fused epilogues,
// async-to-LDS staged local-window attention, wave32 LayerNorm.
// ---------------------------------------------------------------------------

typedef __attribute__((ext_vector_type(16))) _Float16 v16h;
typedef __attribute__((ext_vector_type(8)))  _Float16 v8h;
typedef __attribute__((ext_vector_type(8)))  float    v8f;

#define D_MODEL 512
#define D_FF    2048
#define N_HEADS 8
#define D_HEAD  64
#define BT      8192          // B*T
#define T_LEN   512
#define HALF_W  15
#define CTX_FR  10
#define D_MOT   135

// ============================= GEMM =========================================
// out[M,N] = act( A[M,K] @ W[K,N] + bias[N] ) (+ res[M,N])
// Tile: 64x64 per 256-thread block, K step 32, v_wmma_f32_16x16x32_f16.
#define TM 64
#define TN 64
#define TK 32
#define LDA 48   // halves; row stride 96B (multiple of 16B for ds_load_b128)
#define LDB 48

__global__ __launch_bounds__(256) void cdna5_gemm_f16wmma(
    const float* __restrict__ A, const float* __restrict__ W,
    const float* __restrict__ bias, const float* __restrict__ res,
    float* __restrict__ out, int M, int N, int K, int act,
    const float* __restrict__ alpha_p)
{
  __shared__ __align__(16) _Float16 sA[TM * LDA];
  __shared__ __align__(16) _Float16 sB[TN * LDB];

  const int tid  = threadIdx.x;
  const int m0   = blockIdx.x * TM;
  const int n0   = blockIdx.y * TN;
  const int lane = tid & 31;
  const int wave = tid >> 5;
  const int wm   = wave & 3;   // which 16-row strip
  const int wn   = wave >> 2;  // which 32-col strip
  const int r    = lane & 15;
  const int hi   = lane >> 4;

  // fast path legality is uniform across the block (alignment of row strides)
  const bool fastAB = ((K & 3) == 0) && ((N & 3) == 0);

  v8f acc0 = {};
  v8f acc1 = {};

  for (int k0 = 0; k0 < K; k0 += TK) {
    const bool full = fastAB && (k0 + TK <= K) && (n0 + TN <= N);
    if (full) {
      // ---- A: 64x32, vectorized: 2x float4 load -> 1x ds_store_b128 ----
      {
        int row = tid >> 2, c8 = (tid & 3) * 8;
        const float* ap = A + (size_t)(m0 + row) * K + k0 + c8;
        float4 f0 = ((const float4*)ap)[0];
        float4 f1 = ((const float4*)ap)[1];
        v8h hv;
        hv[0] = (_Float16)f0.x; hv[1] = (_Float16)f0.y;
        hv[2] = (_Float16)f0.z; hv[3] = (_Float16)f0.w;
        hv[4] = (_Float16)f1.x; hv[5] = (_Float16)f1.y;
        hv[6] = (_Float16)f1.z; hv[7] = (_Float16)f1.w;
        *(v8h*)(sA + row * LDA + c8) = hv;
      }
      // ---- B: 32x64 transposed into sB[n][k] ----
      {
        int kk = tid >> 3, n8 = (tid & 7) * 8;
        const float* wp = W + (size_t)(k0 + kk) * N + n0 + n8;
        float4 f0 = ((const float4*)wp)[0];
        float4 f1 = ((const float4*)wp)[1];
        _Float16 hv[8] = {(_Float16)f0.x, (_Float16)f0.y, (_Float16)f0.z,
                          (_Float16)f0.w, (_Float16)f1.x, (_Float16)f1.y,
                          (_Float16)f1.z, (_Float16)f1.w};
#pragma unroll
        for (int j = 0; j < 8; ++j) sB[(n8 + j) * LDB + kk] = hv[j];
      }
    } else {
      // ---- predicated scalar staging (K/N edge tiles only) ----
#pragma unroll
      for (int u = 0; u < 8; ++u) {
        int e   = u * 256 + tid;
        int row = e >> 5, col = e & 31;
        int gk  = k0 + col;
        float v = (gk < K) ? A[(size_t)(m0 + row) * K + gk] : 0.f;
        sA[row * LDA + col] = (_Float16)v;
      }
#pragma unroll
      for (int u = 0; u < 8; ++u) {
        int e  = u * 256 + tid;
        int kk = e >> 6, nn = e & 63;
        int gk = k0 + kk, gn = n0 + nn;
        float v = (gk < K && gn < N) ? W[(size_t)gk * N + gn] : 0.f;
        sB[nn * LDB + kk] = (_Float16)v;
      }
    }
    // prefetch next K tiles while we compute (global_prefetch_b8)
    if (k0 + TK < K) {
      __builtin_prefetch(&A[(size_t)(m0 + (tid >> 2)) * K + (k0 + TK)], 0, 1);
      __builtin_prefetch(&W[(size_t)(k0 + TK + (tid >> 3)) * N + n0], 0, 1);
    }
    __syncthreads();

    // ---- A fragment: 16x32, contiguous k order (k = 16*hi + s), same as B ----
    // lane<16 : h0..15 = K0..15
    // lane>=16: h0..15 = K16..31
    {
      int abase = (wm * 16 + r) * LDA + hi * 16;
      v8h alo = *(const v8h*)(sA + abase);
      v8h ahi = *(const v8h*)(sA + abase + 8);
      v16h af = __builtin_shufflevector(alo, ahi, 0,1,2,3,4,5,6,7,8,9,10,11,12,13,14,15);

      // ---- B fragments: 32x16; lane<16 -> K0..15, lane>=16 -> K16..31 ----
      int bbase0 = (wn * 32 + r) * LDB + hi * 16;
      v8h b0lo = *(const v8h*)(sB + bbase0);
      v8h b0hi = *(const v8h*)(sB + bbase0 + 8);
      v16h bf0 = __builtin_shufflevector(b0lo, b0hi, 0,1,2,3,4,5,6,7,8,9,10,11,12,13,14,15);

      int bbase1 = (wn * 32 + 16 + r) * LDB + hi * 16;
      v8h b1lo = *(const v8h*)(sB + bbase1);
      v8h b1hi = *(const v8h*)(sB + bbase1 + 8);
      v16h bf1 = __builtin_shufflevector(b1lo, b1hi, 0,1,2,3,4,5,6,7,8,9,10,11,12,13,14,15);

      acc0 = __builtin_amdgcn_wmma_f32_16x16x32_f16(false, af, false, bf0,
                                                    (short)0, acc0, false, false);
      acc1 = __builtin_amdgcn_wmma_f32_16x16x32_f16(false, af, false, bf1,
                                                    (short)0, acc1, false, false);
    }
    __syncthreads();
  }

  // ---- epilogue: C/D layout (lane<16: M=r, N=lane; lane>=16: M=r+8) ----
  float alpha = (act == 1 && alpha_p) ? alpha_p[0] : 0.f;
  int gm_base = m0 + wm * 16 + hi * 8;
#pragma unroll
  for (int t = 0; t < 2; ++t) {
    v8f acc = t ? acc1 : acc0;
    int gn = n0 + wn * 32 + t * 16 + r;
    if (gn < N) {
      float bv = bias ? bias[gn] : 0.f;
#pragma unroll
      for (int rr = 0; rr < 8; ++rr) {
        int gm  = gm_base + rr;
        float v = acc[rr] + bv;
        if (act == 1)      v = v > 0.f ? v : alpha * v;   // PReLU
        else if (act == 2) v = v > 0.f ? v : 0.f;          // ReLU
        size_t idx = (size_t)gm * N + gn;
        if (res) v += res[idx];
        out[idx] = v;
      }
    }
  }
}

// =========================== LayerNorm ======================================
// One wave32 per 512-wide row; 8 rows per 256-thread block.
__global__ __launch_bounds__(256) void cdna5_ln(
    const float* __restrict__ x, const float* __restrict__ g,
    const float* __restrict__ b, float* __restrict__ y, int rows)
{
  int wave = threadIdx.x >> 5, lane = threadIdx.x & 31;
  int row = blockIdx.x * 8 + wave;
  if (row >= rows) return;
  const float4* xr = (const float4*)(x + (size_t)row * D_MODEL);
  float4 v[4];
  float s = 0.f;
#pragma unroll
  for (int i = 0; i < 4; ++i) {
    v[i] = xr[lane + i * 32];
    s += v[i].x + v[i].y + v[i].z + v[i].w;
  }
#pragma unroll
  for (int off = 16; off; off >>= 1) s += __shfl_xor(s, off, 32);
  float mu = s * (1.f / 512.f);
  float q = 0.f;
#pragma unroll
  for (int i = 0; i < 4; ++i) {
    float dx = v[i].x - mu; q += dx * dx;
    dx = v[i].y - mu; q += dx * dx;
    dx = v[i].z - mu; q += dx * dx;
    dx = v[i].w - mu; q += dx * dx;
  }
#pragma unroll
  for (int off = 16; off; off >>= 1) q += __shfl_xor(q, off, 32);
  float rs = rsqrtf(q * (1.f / 512.f) + 1e-5f);
  float4* yr = (float4*)(y + (size_t)row * D_MODEL);
#pragma unroll
  for (int i = 0; i < 4; ++i) {
    int c4 = lane + i * 32;
    const float4 gg = ((const float4*)g)[c4];
    const float4 bb = ((const float4*)b)[c4];
    float4 o;
    o.x = (v[i].x - mu) * rs * gg.x + bb.x;
    o.y = (v[i].y - mu) * rs * gg.y + bb.y;
    o.z = (v[i].z - mu) * rs * gg.z + bb.z;
    o.w = (v[i].w - mu) * rs * gg.w + bb.w;
    yr[c4] = o;
  }
}

// ===================== masked motion mix ====================================
__global__ __launch_bounds__(256) void cdna5_maskmix(
    const float* __restrict__ motion, const float* __restrict__ z,
    float* __restrict__ out, int n)
{
  int idx = blockIdx.x * 256 + threadIdx.x;
  if (idx >= n) return;
  int t = (idx / D_MOT) & (T_LEN - 1);
  bool keep = (t < CTX_FR) || (t == T_LEN - 1);
  out[idx] = keep ? motion[idx] : z[idx];
}

// ===================== relative-position embedding ==========================
// rel[31][64] = (prelu(rp*w1 + b1, a1)) @ w2 + b2 ; rp = row-15
__global__ __launch_bounds__(64) void cdna5_relemb(
    const float* __restrict__ w1, const float* __restrict__ b1,
    const float* __restrict__ a1, const float* __restrict__ w2,
    const float* __restrict__ b2, float* __restrict__ rel)
{
  __shared__ float h[D_MODEL];
  int row = blockIdx.x;                 // 0..30
  float rp = (float)(row - HALF_W);
  float alpha = a1[0];
  for (int c = threadIdx.x; c < D_MODEL; c += 64) {
    float v = rp * w1[c] + b1[c];
    h[c] = v > 0.f ? v : alpha * v;
  }
  __syncthreads();
  int n = threadIdx.x;                  // 0..63
  float acc = b2[n];
  for (int c = 0; c < D_MODEL; ++c) acc += h[c] * w2[c * D_HEAD + n];
  rel[row * D_HEAD + n] = acc;
}

// ===================== local-window attention ===============================
// Block = 256 threads = one (b, h, 64-query tile). K/V head-slices for the
// union window [i0-15, i0+78] are staged into LDS with CDNA5 async copies
// (global_load_async_to_lds_b128 + s_wait_asynccnt). 4 threads per query:
// each owns 16 dims; quad-wide __shfl_xor reduces the q.k dot products.
#define AQ    64
#define AROWS (AQ + 2 * HALF_W)     // 94
#define ALD   72                    // float stride (pads away bank conflicts)
#define NWIN  (2 * HALF_W + 1)      // 31

__global__ __launch_bounds__(256) void cdna5_attn(
    const float* __restrict__ Q, const float* __restrict__ K,
    const float* __restrict__ V, const float* __restrict__ rel,
    float* __restrict__ O)
{
  __shared__ __align__(16) float sK[AROWS * ALD];
  __shared__ __align__(16) float sV[AROWS * ALD];
  __shared__ __align__(16) float sR[NWIN * D_HEAD];

  const int tid = threadIdx.x;
  const int bid = blockIdx.x;
  const int b   = bid >> 6;
  const int h   = (bid >> 3) & (N_HEADS - 1);
  const int i0  = (bid & 7) * AQ;

  const unsigned sKo = (unsigned)(size_t)(void*)sK;   // LDS byte offsets
  const unsigned sVo = (unsigned)(size_t)(void*)sV;

  // ---- async stage K/V windows: 94 rows x 64 floats, 16B chunks ----
  const int j0 = i0 - HALF_W;
#pragma unroll
  for (int u = 0; u < 6; ++u) {
    int idx = u * 256 + tid;
    if (idx < AROWS * 16) {
      int rrow = idx >> 4, seg = idx & 15;
      int j = j0 + rrow;
      j = j < 0 ? 0 : (j > T_LEN - 1 ? T_LEN - 1 : j);   // clamped rows are
      size_t goff = (size_t)(b * T_LEN + j) * D_MODEL +  // masked by -1e30
                    h * D_HEAD + seg * 4;
      unsigned lo = (unsigned)((rrow * ALD + seg * 4) * sizeof(float));
      const float* gk = K + goff;
      const float* gv = V + goff;
      asm volatile("global_load_async_to_lds_b128 %0, %1, off"
                   :: "v"(sKo + lo), "v"(gk) : "memory");
      asm volatile("global_load_async_to_lds_b128 %0, %1, off"
                   :: "v"(sVo + lo), "v"(gv) : "memory");
    }
  }
  // rel-bias table (31x64 f32)
  for (int idx = tid; idx < NWIN * D_HEAD; idx += 256) sR[idx] = rel[idx];
  asm volatile("s_wait_asynccnt 0" ::: "memory");
  __syncthreads();

  const int qi   = tid >> 2;          // query within tile
  const int part = tid & 3;           // 16-dim slice
  const int i    = i0 + qi;
  const float* qp = Q + ((size_t)(b * T_LEN + i) * D_MODEL + h * D_HEAD + part * 16);
  float q[16];
#pragma unroll
  for (int d4 = 0; d4 < 4; ++d4) ((float4*)q)[d4] = ((const float4*)qp)[d4];

  float s[NWIN];
  float mx = -1e30f;
#pragma unroll
  for (int jj = 0; jj < NWIN; ++jj) {
    int j = i - HALF_W + jj;
    const float* kr = sK + (qi + jj) * ALD + part * 16;
    const float* rr = sR + jj * D_HEAD + part * 16;
    float p = 0.f;
#pragma unroll
    for (int d = 0; d < 16; ++d) p += q[d] * (kr[d] + rr[d]);
    p += __shfl_xor(p, 1, 32);        // quad reduction
    p += __shfl_xor(p, 2, 32);
    bool valid = (unsigned)j < (unsigned)T_LEN;
    p = valid ? p * 0.125f : -1e30f;  // 1/sqrt(64) scale + window mask
    s[jj] = p;
    mx = fmaxf(mx, p);
  }
  float denom = 0.f;
#pragma unroll
  for (int jj = 0; jj < NWIN; ++jj) { s[jj] = __expf(s[jj] - mx); denom += s[jj]; }
  float inv = 1.f / denom;

  float o[16];
#pragma unroll
  for (int d = 0; d < 16; ++d) o[d] = 0.f;
#pragma unroll
  for (int jj = 0; jj < NWIN; ++jj) {
    const float* vr = sV + (qi + jj) * ALD + part * 16;
    float p = s[jj] * inv;
#pragma unroll
    for (int d = 0; d < 16; ++d) o[d] += p * vr[d];
  }
  float* op = O + ((size_t)(b * T_LEN + i) * D_MODEL + h * D_HEAD + part * 16);
#pragma unroll
  for (int d4 = 0; d4 < 4; ++d4) ((float4*)op)[d4] = ((float4*)o)[d4];
}

// =============================== host =======================================

extern "C" void kernel_launch(void* const* d_in, const int* in_sizes, int n_in,
                              void* d_out, int out_size, void* d_ws, size_t ws_size,
                              hipStream_t stream) {
  (void)in_sizes; (void)n_in; (void)out_size; (void)ws_size;

  // --- input leaf indices (depth-first, insertion order) ---
  // 0 motion, 1 traj, 2 z
  // motion_enc: 3 l1.w, 4 l1.b, 5 a1, 6 l2.w, 7 l2.b, 8 a2
  // traj_enc:   9..14 (same pattern)
  // rel_enc:    15 l1.w, 16 l1.b, 17 a1, 18 l2.w, 19 l2.b
  // layers: base 20 + 26*l
  //   self:  +0 q.w,+1 q.b,+2 k.w,+3 k.b,+4 v.w,+5 v.b,+6 o.w,+7 o.b,+8 ln.g,+9 ln.b
  //   cross: +10..+19 same
  //   pffn:  +20 l1.w,+21 l1.b,+22 l2.w,+23 l2.b,+24 ln.g,+25 ln.b
  // final_ln: 176 g, 177 b ; decoder: 178 l1.w, 179 l1.b, 180 a, 181 l2.w, 182 l2.b
  const float* motion = (const float*)d_in[0];
  const float* traj   = (const float*)d_in[1];
  const float* z      = (const float*)d_in[2];
  auto P = [&](int i) { return (const float*)d_in[i]; };

  // --- workspace layout (floats) ---
  float* ws = (float*)d_ws;
  float* rel = ws;                                   // 31*64 -> pad 2048
  size_t off = 2048;
  float* mm  = ws + off; off += (size_t)BT * D_MOT;  // masked motion
  float* x   = ws + off; off += (size_t)BT * D_MODEL;
  float* ctx = ws + off; off += (size_t)BT * D_MODEL;
  float* xn  = ws + off; off += (size_t)BT * D_MODEL;
  float* cn  = ws + off; off += (size_t)BT * D_MODEL;
  float* big = ws + off;                             // BT * D_FF region
  float* qb = big;
  float* kb = big + (size_t)BT * D_MODEL;
  float* vb = big + (size_t)2 * BT * D_MODEL;
  float* ao = big + (size_t)3 * BT * D_MODEL;
  float* hb = big;                                   // FFN hidden reuses region

  dim3 blk(256);
  auto gemm = [&](const float* A, int widx, const float* res, float* out,
                  int M, int N, int K, int act, int aidx) {
    dim3 grid(M / TM, (N + TN - 1) / TN);
    hipLaunchKernelGGL(cdna5_gemm_f16wmma, grid, blk, 0, stream,
                       A, P(widx), P(widx + 1), res, out, M, N, K, act,
                       aidx >= 0 ? P(aidx) : nullptr);
  };
  auto ln = [&](const float* in, int gidx, float* out) {
    hipLaunchKernelGGL(cdna5_ln, dim3(BT / 8), blk, 0, stream,
                       in, P(gidx), P(gidx + 1), out, BT);
  };
  auto attn = [&](const float* q, const float* k, const float* v, float* o) {
    hipLaunchKernelGGL(cdna5_attn, dim3(16 * N_HEADS * (T_LEN / AQ)), blk, 0,
                       stream, q, k, v, rel, o);
  };

  // 1) masked motion mix
  {
    int n = BT * D_MOT;
    hipLaunchKernelGGL(cdna5_maskmix, dim3((n + 255) / 256), blk, 0, stream,
                       motion, z, mm, n);
  }
  // 2) motion encoder -> x
  gemm(mm, 3, nullptr, xn, BT, D_MODEL, D_MOT, 1, 5);
  gemm(xn, 6, nullptr, x,  BT, D_MODEL, D_MODEL, 1, 8);
  // 3) traj encoder -> ctx
  gemm(traj, 9, nullptr, cn,  BT, D_MODEL, 3, 1, 11);
  gemm(cn,  12, nullptr, ctx, BT, D_MODEL, D_MODEL, 1, 14);
  // 4) relative position table
  hipLaunchKernelGGL(cdna5_relemb, dim3(NWIN), dim3(64), 0, stream,
                     P(15), P(16), P(17), P(18), P(19), rel);

  // 5) transformer layers
  for (int l = 0; l < 6; ++l) {
    int L = 20 + 26 * l;
    // -- self attention --
    ln(x, L + 8, xn);
    gemm(xn, L + 0, nullptr, qb, BT, D_MODEL, D_MODEL, 0, -1);
    gemm(xn, L + 2, nullptr, kb, BT, D_MODEL, D_MODEL, 0, -1);
    gemm(xn, L + 4, nullptr, vb, BT, D_MODEL, D_MODEL, 0, -1);
    attn(qb, kb, vb, ao);
    gemm(ao, L + 6, x, x, BT, D_MODEL, D_MODEL, 0, -1);
    // -- cross attention --
    ln(x,   L + 18, xn);
    ln(ctx, L + 18, cn);
    gemm(xn, L + 10, nullptr, qb, BT, D_MODEL, D_MODEL, 0, -1);
    gemm(cn, L + 12, nullptr, kb, BT, D_MODEL, D_MODEL, 0, -1);
    gemm(cn, L + 14, nullptr, vb, BT, D_MODEL, D_MODEL, 0, -1);
    attn(qb, kb, vb, ao);
    gemm(ao, L + 16, x, x, BT, D_MODEL, D_MODEL, 0, -1);
    // -- pffn --
    ln(x, L + 24, xn);
    gemm(xn, L + 20, nullptr, hb, BT, D_FF, D_MODEL, 2, -1);
    gemm(hb, L + 22, x, x, BT, D_MODEL, D_FF, 0, -1);
  }

  // 6) final LN + decoder
  ln(x, 176, xn);
  gemm(xn, 178, nullptr, cn, BT, D_MODEL, D_MODEL, 1, 180);
  gemm(cn, 181, nullptr, (float*)d_out, BT, D_MOT, D_MODEL, 0, -1);
}
